// StandardAttention_3659312136327
// MI455X (gfx1250) — matrix operand order repaired and checked by reading, not compiled, not run on hardware
//
#include <hip/hip_runtime.h>

// ---------------------------------------------------------------- constants
#define B_  2
#define N_  4096
#define D_  512
#define H_  8
#define HD_ 64
#define NQKV_ (3 * D_)   // 1536

typedef _Float16 half_t;
typedef __attribute__((ext_vector_type(16))) _Float16 v16h;
typedef __attribute__((ext_vector_type(8)))  float    v8f;

union Frag16 {
    v16h  h;
    uint4 q[2];
};

// A-fragment (16x32 f16, M x K): lane (l,hi): halves j=0..7 at K=k0+hi*8+j,
// halves j=8..15 at K=k0+16+hi*8+(j-8).  (ISA 7.12.2, 16-bit A 16x32)
__device__ __forceinline__ v16h load_frag_a(const half_t* base, int strideH,
                                            int row, int hi, int k0) {
    const half_t* r = base + (size_t)row * strideH + k0 + hi * 8;
    Frag16 f;
    f.q[0] = *(const uint4*)(r);
    f.q[1] = *(const uint4*)(r + 16);
    return f.h;
}

// B-fragment (32x16 f16, K x N): lane (l,hi) = column N=l: halves j=0..7 at
// K=k0+hi*8+j, halves j=8..15 at K=k0+16+hi*8+(j-8) (same order as A).
__device__ __forceinline__ v16h load_frag_b(const half_t* base, int strideH,
                                            int row, int hi, int k0) {
    const half_t* r = base + (size_t)row * strideH + k0 + hi * 8;
    Frag16 f;
    f.q[0] = *(const uint4*)(r);
    f.q[1] = *(const uint4*)(r + 16);
    return f.h;
}

__device__ __forceinline__ v8f wmma_f16(v16h a, v16h b, v8f c) {
    return __builtin_amdgcn_wmma_f32_16x16x32_f16(
        /*neg_a=*/false, a, /*neg_b=*/false, b,
        /*c_mod=*/(short)0, c, /*reuse_a=*/false, /*reuse_b=*/false);
}

// gfx1250 async memory->LDS copy, 16 bytes per lane, tracked by ASYNCcnt.
// VDST carries the per-lane LDS byte offset (low 32 bits of a generic LDS
// pointer ARE the LDS offset on AMDGPU); VADDR is the 64-bit global address.
__device__ __forceinline__ void async_copy16(half_t* lds_dst,
                                             const half_t* gsrc) {
    uint32_t lds = (uint32_t)(uintptr_t)lds_dst;
    uint64_t ga = (uint64_t)(uintptr_t)gsrc;
    asm volatile("global_load_async_to_lds_b128 %0, %1, off"
                 :: "v"(lds), "v"(ga)
                 : "memory");
}
// Wait until at most `n` async ops remain outstanding (they retire in order,
// so waiting <= N while the newest batch of N is in flight completes the
// previous batch; this is what makes the double-buffer pipeline work).
#define WAIT_ASYNC(n) asm volatile("s_wait_asynccnt " #n ::: "memory")

// reduce across a 16-lane group (rows of a C tile live across 16 lanes)
__device__ __forceinline__ float red_max16(float x) {
#pragma unroll
    for (int off = 8; off; off >>= 1) x = fmaxf(x, __shfl_xor(x, off, 32));
    return x;
}
__device__ __forceinline__ float red_sum16(float x) {
#pragma unroll
    for (int off = 8; off; off >>= 1) x += __shfl_xor(x, off, 32);
    return x;
}

// ---------------------------------------------------------------- convert
__global__ void f32_to_f16_kernel(const float* __restrict__ s,
                                  half_t* __restrict__ d, int n) {
    for (int i = blockIdx.x * blockDim.x + threadIdx.x; i < n;
         i += gridDim.x * blockDim.x)
        d[i] = (half_t)s[i];
}

// ------------------------------------------------- shared GEMM mainloop
// 64x64 block tile, 4 waves (16 rows each), K chunks of 32, double-buffered
// LDS with async A staging pipelined one chunk ahead of the WMMA phase.
__device__ __forceinline__ void gemm64_mainloop(
    const half_t* __restrict__ A, const half_t* __restrict__ W, int ldw,
    int m0, int n0, int tid, int wv, int l, int hi,
    half_t (&As)[2][64 * 40], half_t (&Bs)[2][64 * 40], v8f (&c)[4]) {
    // stage chunk kc into buffer buf: async A rows + W->Bs transpose
    auto stage = [&](int kc, int buf) {
#pragma unroll
        for (int i = 0; i < 2; ++i) {
            int idx = tid + i * 128;
            int row = idx >> 2, c4 = idx & 3;
            async_copy16(&As[buf][row * 40 + c4 * 8],
                         A + (size_t)(m0 + row) * D_ + kc + c4 * 8);
        }
#pragma unroll
        for (int i = 0; i < 2; ++i) {
            int idx = tid + i * 128;            // 256 chunks of 8 halves
            int k = idx >> 3, n8 = (idx & 7) * 8;
            union { uint4 q; half_t h[8]; } t;
            t.q = *(const uint4*)(W + (size_t)(kc + k) * ldw + n0 + n8);
#pragma unroll
            for (int j = 0; j < 8; ++j) Bs[buf][(n8 + j) * 40 + k] = t.h[j];
        }
    };

    stage(0, 0);
    for (int t = 0; t < D_ / 32; ++t) {
        int cur = t & 1;
        if (t < D_ / 32 - 1) {
            stage((t + 1) * 32, cur ^ 1);  // next chunk streams during WMMA
            WAIT_ASYNC(2);                 // current chunk's A is complete
        } else {
            WAIT_ASYNC(0);
        }
        __syncthreads();

        v16h af = load_frag_a(&As[cur][0], 40, wv * 16 + l, hi, 0);
#pragma unroll
        for (int ns = 0; ns < 4; ++ns) {
            v16h bf = load_frag_b(&Bs[cur][0], 40, ns * 16 + l, hi, 0);
            c[ns] = wmma_f16(af, bf, c[ns]);
        }
        __syncthreads();   // readers done before buffer is re-staged
    }
}

// ---------------------------------------------------------------- QKV GEMM
// X[8192,512](f16) @ Wqkv[512,1536](f16) + b -> scatter Q,K as [b,h,n,hd],
// V transposed as [b,h,hd,n].
__global__ __launch_bounds__(128) void qkv_gemm_kernel(
    const half_t* __restrict__ A, const half_t* __restrict__ W,
    const float* __restrict__ bias, half_t* __restrict__ Qg,
    half_t* __restrict__ Kg, half_t* __restrict__ Vg) {
    __shared__ __align__(16) half_t As[2][64 * 40];
    __shared__ __align__(16) half_t Bs[2][64 * 40];

    const int tid = threadIdx.x;
    const int wv = tid >> 5, lane = tid & 31, l = lane & 15, hi = lane >> 4;
    const int m0 = blockIdx.x * 64;
    const int n0 = blockIdx.y * 64;

    v8f c[4] = {};
    gemm64_mainloop(A, W, NQKV_, m0, n0, tid, wv, l, hi, As, Bs, c);

    // bias add + scatter: destination differs only by base pointer and row
    // stride (uniform per block since n0 is 64-aligned within one matrix).
#pragma unroll
    for (int ns = 0; ns < 4; ++ns) {
        int col = n0 + ns * 16 + l;
        float bv = bias[col];
        int mat = col >> 9;            // 0:Q 1:K 2:V (uniform per block)
        int hh = (col >> 6) & 7;
        int hd = col & 63;
        int grow0 = m0 + wv * 16 + hi * 8;        // 16-row slab never crosses
        int bb = grow0 >> 12, n = grow0 & (N_ - 1);  // the batch boundary
        size_t bhI = (size_t)(bb * H_ + hh);
        half_t* dst;
        size_t base, stride;
        if (mat == 2) {
            dst = Vg; base = (bhI * HD_ + hd) * N_ + n; stride = 1;
        } else {
            dst = (mat == 0) ? Qg : Kg;
            base = (bhI * N_ + n) * HD_ + hd; stride = HD_;
        }
#pragma unroll
        for (int r = 0; r < 8; ++r)
            dst[base + (size_t)r * stride] = (half_t)(c[ns][r] + bv);
    }
}

// ---------------------------------------------------------------- attention
// One block per (b, h, 64-query tile); 4 waves, each owns 16 query rows.
// Flash attention, kt <= qt (causal tile skipping).  K/V tiles staged with
// gfx1250 async-to-LDS, double-buffered one tile ahead of the WMMA phase.
__global__ __launch_bounds__(128) void attn_kernel(
    const half_t* __restrict__ Qg, const half_t* __restrict__ Kg,
    const half_t* __restrict__ Vg, half_t* __restrict__ Og) {
    __shared__ __align__(16) half_t Ks[2][64 * 72];    // [key][hd], pad 8
    __shared__ __align__(16) half_t Vts[2][64 * 72];   // [hd][key], pad 8
    __shared__ __align__(16) half_t Ps[4 * 16 * 72];   // per-wave P patch

    const int tid = threadIdx.x;
    const int wv = tid >> 5, lane = tid & 31, l = lane & 15, hi = lane >> 4;

    const int bid = blockIdx.x;
    const int qt = bid & 63;            // N/64 query tiles
    const int hh = (bid >> 6) & 7;
    const int bb = bid >> 9;

    const int m0 = qt * 64 + wv * 16;   // first query row of this wave
    const size_t bh = (size_t)(bb * H_ + hh);
    const half_t* Qb = Qg + bh * (size_t)N_ * HD_;
    const half_t* Kb = Kg + bh * (size_t)N_ * HD_;
    const half_t* Vb = Vg + bh * (size_t)HD_ * N_;

    // Q fragments (16 rows x 64 hd = two 16x32 A fragments), in registers
    v16h qf[2];
#pragma unroll
    for (int kc = 0; kc < 2; ++kc)
        qf[kc] = load_frag_a(Qb + (size_t)m0 * HD_, HD_, l, hi, kc * 32);

    v8f o[4] = {};
    float mrow[8], lrow[8];
#pragma unroll
    for (int r = 0; r < 8; ++r) { mrow[r] = -1.0e30f; lrow[r] = 0.0f; }

    half_t* Pw = &Ps[wv * 16 * 72];

    // async-stage one 64-key tile (8 b128 per thread => ASYNCcnt batch of 8)
    auto stage_kv = [&](int kt, int buf) {
#pragma unroll
        for (int i = 0; i < 4; ++i) {
            int idx = tid + i * 128;
            int row = idx >> 3, c8 = idx & 7;
            async_copy16(&Ks[buf][row * 72 + c8 * 8],
                         Kb + (size_t)(kt * 64 + row) * HD_ + c8 * 8);
            async_copy16(&Vts[buf][row * 72 + c8 * 8],
                         Vb + (size_t)row * N_ + kt * 64 + c8 * 8);
        }
    };

    stage_kv(0, 0);
    for (int kt = 0; kt <= qt; ++kt) {
        const int cur = kt & 1;
        if (kt < qt) {
            stage_kv(kt + 1, cur ^ 1);   // next tile streams during compute
            WAIT_ASYNC(8);               // current tile is complete
        } else {
            WAIT_ASYNC(0);
        }
        if (kt + 2 <= qt) {              // L2 prefetch two tiles ahead
            __builtin_prefetch(Kb + (size_t)(kt + 2) * 64 * HD_, 0, 1);
            __builtin_prefetch(Vb + (kt + 2) * 64, 0, 1);
        }
        __syncthreads();

        // S = Q @ K^T : 4 key slabs x 2 hd chunks = 8 WMMA
        v8f s[4];
#pragma unroll
        for (int ns = 0; ns < 4; ++ns) {
            v8f acc = {};
#pragma unroll
            for (int kc = 0; kc < 2; ++kc) {
                v16h bf = load_frag_b(&Ks[cur][0], 72, ns * 16 + l, hi, kc * 32);
                acc = wmma_f16(qf[kc], bf, acc);
            }
            s[ns] = acc;
        }

        // scale + causal mask (only the diagonal tile needs masking)
        float rmax[8];
#pragma unroll
        for (int r = 0; r < 8; ++r) {
            float mx = -1.0e30f;
#pragma unroll
            for (int ns = 0; ns < 4; ++ns) {
                float v = s[ns][r] * 0.125f;   // hd^-0.5
                if (kt == qt) {
                    int key = kt * 64 + ns * 16 + l;
                    int qr = m0 + r + hi * 8;
                    if (key > qr) v = -1.0e30f;
                }
                s[ns][r] = v;
                mx = fmaxf(mx, v);
            }
            rmax[r] = red_max16(mx);
        }

        // online softmax update
#pragma unroll
        for (int r = 0; r < 8; ++r) {
            float mnew = fmaxf(mrow[r], rmax[r]);
            float alpha = __expf(mrow[r] - mnew);
            mrow[r] = mnew;
            float rs = 0.0f;
#pragma unroll
            for (int ns = 0; ns < 4; ++ns) {
                float p = __expf(s[ns][r] - mnew);
                s[ns][r] = p;
                rs += p;
            }
            lrow[r] = lrow[r] * alpha + red_sum16(rs);
#pragma unroll
            for (int ns = 0; ns < 4; ++ns) o[ns][r] *= alpha;
        }

        // route P: C-layout -> LDS -> A-layout (per-wave patch; DS is
        // in-order within a wave, so only a compiler fence is needed)
#pragma unroll
        for (int ns = 0; ns < 4; ++ns)
#pragma unroll
            for (int r = 0; r < 8; ++r)
                Pw[(r + hi * 8) * 72 + ns * 16 + l] = (half_t)s[ns][r];
        __asm volatile("" ::: "memory");

        // O += P @ V : 2 key chunks x 4 hd slabs = 8 WMMA
#pragma unroll
        for (int kc = 0; kc < 2; ++kc) {
            v16h pa = load_frag_a(Pw, 72, l, hi, kc * 32);
#pragma unroll
            for (int ns = 0; ns < 4; ++ns) {
                v16h vb = load_frag_b(&Vts[cur][0], 72, ns * 16 + l, hi, kc * 32);
                o[ns] = wmma_f16(pa, vb, o[ns]);
            }
        }
        __syncthreads();   // readers done before buffer is re-staged
    }

    // normalize and store O as f16 [B*N, D] row-major (ready for out GEMM)
    const size_t orow0 = (size_t)bb * N_ + m0;
#pragma unroll
    for (int ns = 0; ns < 4; ++ns)
#pragma unroll
        for (int r = 0; r < 8; ++r) {
            float v = o[ns][r] / lrow[r];
            Og[(orow0 + r + hi * 8) * D_ + hh * HD_ + ns * 16 + l] = (half_t)v;
        }
}

// ---------------------------------------------------------------- out GEMM
// O[8192,512](f16) @ Wout[512,512](f16) + b_out -> d_out fp32
__global__ __launch_bounds__(128) void out_gemm_kernel(
    const half_t* __restrict__ A, const half_t* __restrict__ W,
    const float* __restrict__ bias, float* __restrict__ Out) {
    __shared__ __align__(16) half_t As[2][64 * 40];
    __shared__ __align__(16) half_t Bs[2][64 * 40];

    const int tid = threadIdx.x;
    const int wv = tid >> 5, lane = tid & 31, l = lane & 15, hi = lane >> 4;
    const int m0 = blockIdx.x * 64;
    const int n0 = blockIdx.y * 64;

    v8f c[4] = {};
    gemm64_mainloop(A, W, D_, m0, n0, tid, wv, l, hi, As, Bs, c);

#pragma unroll
    for (int ns = 0; ns < 4; ++ns) {
        int col = n0 + ns * 16 + l;
        float bv = bias[col];
#pragma unroll
        for (int r = 0; r < 8; ++r) {
            int grow = m0 + wv * 16 + r + hi * 8;
            Out[(size_t)grow * D_ + col] = c[ns][r] + bv;
        }
    }
}

// ---------------------------------------------------------------- launch
extern "C" void kernel_launch(void* const* d_in, const int* in_sizes, int n_in,
                              void* d_out, int out_size, void* d_ws,
                              size_t ws_size, hipStream_t stream) {
    (void)in_sizes; (void)n_in; (void)out_size; (void)ws_size;
    const float* x = (const float*)d_in[0];     // [B,N,D]
    const float* Wqkv = (const float*)d_in[1];  // [D,3D]
    const float* bqkv = (const float*)d_in[2];  // [3D]
    const float* Wout = (const float*)d_in[3];  // [D,D]
    const float* bout = (const float*)d_in[4];  // [D]
    float* out = (float*)d_out;

    char* ws = (char*)d_ws;
    size_t off = 0;
    half_t* xh = (half_t*)(ws + off);  off += (size_t)B_ * N_ * D_ * 2;       // 8.39 MB
    half_t* wqh = (half_t*)(ws + off); off += (size_t)D_ * NQKV_ * 2;         // 1.57 MB
    half_t* woh = (half_t*)(ws + off); off += (size_t)D_ * D_ * 2;            // 0.52 MB
    half_t* Qg = (half_t*)(ws + off);  off += (size_t)B_ * H_ * N_ * HD_ * 2; // 8.39 MB
    half_t* Kg = (half_t*)(ws + off);  off += (size_t)B_ * H_ * N_ * HD_ * 2;
    half_t* Vg = (half_t*)(ws + off);  off += (size_t)B_ * H_ * N_ * HD_ * 2;
    half_t* Og = (half_t*)(ws + off);  off += (size_t)B_ * N_ * D_ * 2;       // total ~44 MB

    f32_to_f16_kernel<<<2048, 256, 0, stream>>>(x, xh, B_ * N_ * D_);
    f32_to_f16_kernel<<<1024, 256, 0, stream>>>(Wqkv, wqh, D_ * NQKV_);
    f32_to_f16_kernel<<<512, 256, 0, stream>>>(Wout, woh, D_ * D_);

    qkv_gemm_kernel<<<dim3((B_ * N_) / 64, NQKV_ / 64), 128, 0, stream>>>(
        xh, wqh, bqkv, Qg, Kg, Vg);

    attn_kernel<<<B_ * H_ * (N_ / 64), 128, 0, stream>>>(Qg, Kg, Vg, Og);

    out_gemm_kernel<<<dim3((B_ * N_) / 64, D_ / 64), 128, 0, stream>>>(
        Og, woh, bout, out);
}
